// MB_projection_55430847922251
// MI455X (gfx1250) — matrix operand order repaired and checked by reading, not compiled, not run on hardware
//
#include <hip/hip_runtime.h>
#include <stdint.h>

typedef __attribute__((ext_vector_type(16))) _Float16 v16h;
typedef __attribute__((ext_vector_type(8)))  float    v8f;

#define IN_F   512
#define OUT_F  10240
#define BATCH  4096
#define KSTEPS (IN_F / 32)        // 16
#define MTILES (BATCH / 16)       // 256
#define NTILES (OUT_F / 16)       // 640

// Workspace layout (bytes):
//   Ahi frags: [MTILES*KSTEPS][32 lanes] v16h   = 4 MB    at offset 0
//   Alo frags: same                              = 4 MB    at offset 4 MB
//   Wh  frags: [NTILES*KSTEPS][32 lanes] v16h   = 10.5 MB at offset 8 MB
#define AHI_OFF  0
#define ALO_OFF  ((size_t)BATCH * IN_F * 2)            // 4194304
#define WH_OFF   (2 * ALO_OFF)                         // 8388608

// ---------------------------------------------------------------------------
// Pack A into f16 hi/lo split, in the WMMA A-fragment (16x32) lane layout:
//   lane<16 : row = l16, K = k0 + {0..7, 16..23}
//   lane>=16: row = l16, K = k0 + {8..15, 24..31}
// One thread builds one lane's v16h => 32B contiguous stores, fully coalesced.
// ---------------------------------------------------------------------------
__global__ __launch_bounds__(256) void pack_a(
    const float* __restrict__ A, v16h* __restrict__ Ahi, v16h* __restrict__ Alo)
{
    const int t    = blockIdx.x * 256 + threadIdx.x;   // 0 .. MTILES*KSTEPS*32-1
    const int lane = t & 31;
    const int frag = t >> 5;                            // mTile*KSTEPS + kstep
    const int mTile = frag >> 4;
    const int kstep = frag & 15;
    const int row  = mTile * 16 + (lane & 15);
    const int hi   = lane >> 4;
    const int kb   = kstep * 32 + hi * 8;
    const float* __restrict__ ar = A + (size_t)row * IN_F + kb;

    v16h ah, al;
    #pragma unroll
    for (int i = 0; i < 16; ++i) {
        const float f = ar[(i < 8) ? i : (i + 8)];      // {0..7, 16..23}
        const _Float16 h = (_Float16)f;
        ah[i] = h;
        al[i] = (_Float16)(f - (float)h);
    }
    Ahi[t] = ah;
    Alo[t] = al;
}

// ---------------------------------------------------------------------------
// Pack W (exact {0,1} -> f16) in the WMMA B-fragment (32x16) lane layout:
//   lane<16 : col = l16, K = k0 + {0..7, 16..23} ; lane>=16: col = l16, K = k0 + {8..15, 24..31}
// ---------------------------------------------------------------------------
__global__ __launch_bounds__(256) void pack_b(
    const float* __restrict__ W, v16h* __restrict__ Wh)
{
    const int t    = blockIdx.x * 256 + threadIdx.x;   // 0 .. NTILES*KSTEPS*32-1
    const int lane = t & 31;
    const int frag = t >> 5;                            // nTile*KSTEPS + kstep
    const int nTile = frag >> 4;
    const int kstep = frag & 15;
    const int col  = nTile * 16 + (lane & 15);
    const int kb   = kstep * 32 + (lane >> 4) * 8;
    const float* __restrict__ wr = W + (size_t)col * IN_F + kb;

    v16h bh;
    #pragma unroll
    for (int i = 0; i < 16; ++i) bh[i] = (_Float16)wr[(i < 8) ? i : (i + 8)];
    Wh[t] = bh;
}

#define WMMA(a, b, c) \
    __builtin_amdgcn_wmma_f32_16x16x32_f16(false, (a), false, (b), (short)0, (c), false, false)

// ---------------------------------------------------------------------------
// GEMM: X[b, o] = sum_k A[b, k] * W[o, k]
// Each wave32 computes a 32(M) x 128(N) tile: 16 f32 accumulator tiles.
// Per k-step: 12 fragment loads (12 KB/wave) feed 32 WMMAs (512 KFLOP)
// => 0.023 B/FLOP of L2 traffic; every B fragment reused 4x from registers.
// hi/lo f16 split of A gives ~fp32-exact results (W is exact in f16).
// ---------------------------------------------------------------------------
__global__ __launch_bounds__(256) void flyhash_gemm(
    const v16h* __restrict__ Ahi, const v16h* __restrict__ Alo,
    const v16h* __restrict__ Wh,  float* __restrict__ X)
{
    const int lane = threadIdx.x & 31;
    const int wave = threadIdx.x >> 5;
    const int l16  = lane & 15;
    const int hi   = lane >> 4;

    const int NSTRIPS = OUT_F / 128;          // 80 strips of 128 columns
    const int wid   = blockIdx.x * 8 + wave;  // 0..10239
    const int mPair = wid / NSTRIPS;          // 0..127 (pair of 16-row M tiles)
    const int nT0   = (wid % NSTRIPS) * 8;    // first of 8 N-tiles

    const v16h* __restrict__ aHi0 = Ahi + ((size_t)(mPair * 2 + 0) * KSTEPS) * 32 + lane;
    const v16h* __restrict__ aLo0 = Alo + ((size_t)(mPair * 2 + 0) * KSTEPS) * 32 + lane;
    const v16h* __restrict__ aHi1 = Ahi + ((size_t)(mPair * 2 + 1) * KSTEPS) * 32 + lane;
    const v16h* __restrict__ aLo1 = Alo + ((size_t)(mPair * 2 + 1) * KSTEPS) * 32 + lane;
    const v16h* __restrict__ bP   = Wh  + ((size_t)nT0 * KSTEPS) * 32 + lane;

    v8f acc0[8] = {}, acc1[8] = {};

    for (int ks = 0; ks < KSTEPS; ++ks) {
        const v16h ah0 = aHi0[ks * 32];
        const v16h al0 = aLo0[ks * 32];
        const v16h ah1 = aHi1[ks * 32];
        const v16h al1 = aLo1[ks * 32];

        #pragma unroll
        for (int t = 0; t < 8; ++t) {
            const v16h b = bP[(t * KSTEPS + ks) * 32];
            acc0[t] = WMMA(ah0, b, acc0[t]);
            acc0[t] = WMMA(al0, b, acc0[t]);
            acc1[t] = WMMA(ah1, b, acc1[t]);
            acc1[t] = WMMA(al1, b, acc1[t]);
        }
    }

    // C/D 16x16 f32 layout: VGPR r -> (lanes 0-15: M=r, N=l16), (lanes 16-31: M=r+8)
    #pragma unroll
    for (int t = 0; t < 8; ++t) {
        const int col = (nT0 + t) * 16 + l16;
        #pragma unroll
        for (int r = 0; r < 8; ++r) {
            const int orow0 = (mPair * 2 + 0) * 16 + r + 8 * hi;
            const int orow1 = (mPair * 2 + 1) * 16 + r + 8 * hi;
            X[(size_t)orow0 * OUT_F + col] = acc0[t][r];
            X[(size_t)orow1 * OUT_F + col] = acc1[t][r];
        }
    }
}

// ---------------------------------------------------------------------------
// Per-row exact k-th-largest via 4-pass radix select on float bit patterns
// (all values >= 0, so uint bit order == float order), then threshold.
// One 256-thread block per row; row staged in LDS (40 KB of 320 KB WGP LDS).
// ---------------------------------------------------------------------------
__global__ __launch_bounds__(256) void topk_threshold(
    float* __restrict__ X, const int* __restrict__ kptr)
{
    __shared__ uint32_t srow[OUT_F];     // 40 KB
    __shared__ uint32_t hist[256];
    __shared__ uint32_t s_prefix;
    __shared__ uint32_t s_remain;

    const int tid = threadIdx.x;
    float* __restrict__ xrow = X + (size_t)blockIdx.x * OUT_F;

    for (int i = tid; i < OUT_F; i += 256)
        srow[i] = __float_as_uint(xrow[i]);
    if (tid == 0) { s_prefix = 0u; s_remain = (uint32_t)(*kptr); }
    __syncthreads();

    for (int p = 3; p >= 0; --p) {
        hist[tid] = 0u;
        __syncthreads();

        const uint32_t km = (p == 3) ? 0u : (0xFFFFFFFFu << ((p + 1) * 8));
        const int      sh = p * 8;
        const uint32_t prefix = s_prefix;

        for (int i = tid; i < OUT_F; i += 256) {
            const uint32_t u = srow[i];
            if ((u & km) == prefix)
                atomicAdd(&hist[(u >> sh) & 0xFFu], 1u);
        }
        __syncthreads();

        if (tid == 0) {
            uint32_t remain = s_remain, cum = 0u;
            for (int b = 255; b >= 0; --b) {
                const uint32_t c = hist[b];
                if (cum + c >= remain) {
                    s_prefix = prefix | ((uint32_t)b << sh);
                    s_remain = remain - cum;
                    break;
                }
                cum += c;
            }
        }
        __syncthreads();
    }

    const float thr = __uint_as_float(s_prefix);   // exact k-th largest
    for (int i = tid; i < OUT_F; i += 256) {
        const float f = __uint_as_float(srow[i]);
        xrow[i] = (f >= thr) ? f : 0.0f;
    }
}

extern "C" void kernel_launch(void* const* d_in, const int* in_sizes, int n_in,
                              void* d_out, int out_size, void* d_ws, size_t ws_size,
                              hipStream_t stream)
{
    const float* inp = (const float*)d_in[0];   // [4096, 512]  f32
    const float* w   = (const float*)d_in[1];   // [10240, 512] f32
    const int*   hl  = (const int*)d_in[2];     // hash_length (device scalar)
    float* X = (float*)d_out;                   // [4096, 10240] f32

    char* ws = (char*)d_ws;
    v16h* Ahi = (v16h*)(ws + AHI_OFF);
    v16h* Alo = (v16h*)(ws + ALO_OFF);
    v16h* Whf = (v16h*)(ws + WH_OFF);

    // 1) one-time conversion into WMMA fragment layouts (coalesced 32B stores)
    pack_a<<<(MTILES * KSTEPS * 32) / 256, 256, 0, stream>>>(inp, Ahi, Alo);
    pack_b<<<(NTILES * KSTEPS * 32) / 256, 256, 0, stream>>>(w, Whf);

    // 2) WMMA GEMM: 10240 wave-tiles (32x128 each), 8 waves per block
    const int waveTiles = (BATCH / 32) * (OUT_F / 128);
    flyhash_gemm<<<waveTiles / 8, 256, 0, stream>>>(Ahi, Alo, Whf, X);

    // 3) exact per-row top-k threshold, in place
    topk_threshold<<<BATCH, 256, 0, stream>>>(X, hl);
}
